// Decoder_30588757082990
// MI455X (gfx1250) — matrix operand order repaired and checked by reading, not compiled, not run on hardware
//
#include <hip/hip_runtime.h>
#include <hip/hip_bf16.h>

// ---------------- problem constants ----------------
#define Bn   128
#define NPn  196
#define FEATn 2048
#define En   512
#define Dn   512
#define Vn   10000
#define Ln   52
#define Tn   51

// ---------------- workspace layout (bytes) ----------------
#define OFF_SORT    ((size_t)0)            // int[128]
#define OFF_DECLEN  ((size_t)512)          // int[128]
#define OFF_CAPS    ((size_t)1024)         // int[128*52]
#define OFF_FMEAN   ((size_t)28672)        // bf16[128*2048]
#define OFF_ATT1W   ((size_t)552960)       // bf16[512*2048]
#define OFF_WIH     ((size_t)2650112)      // bf16[2048*512]
#define OFF_WHH     ((size_t)4747264)      // bf16[2048*512]
#define OFF_WORDW   ((size_t)6844416)      // bf16[10000*512]
#define OFF_EMB     ((size_t)17084416)     // bf16[128*52*512]
#define OFF_H       ((size_t)23900160)     // float[2][128*512]
#define OFF_C       ((size_t)24424448)     // float[2][128*512]
#define OFF_HBF     ((size_t)24948736)     // bf16[2][128*512]
#define OFF_HNEW    ((size_t)25210880)     // bf16[128*512]

typedef __attribute__((ext_vector_type(8)))  __bf16 bf16x8;
typedef __attribute__((ext_vector_type(16))) __bf16 bf16x16;
typedef __attribute__((ext_vector_type(8)))  float  f32x8;

// A fragment: 16x32 bf16. Lanes 0-15 (M=lane) hold K=0..7 then K=16..23,
// lanes 16-31 (M=lane-16) hold K=8..15 then K=24..31.
__device__ __forceinline__ bf16x16 ldA(const __bf16* base, int rowStride,
                                       int m0, int k, int lane) {
    int half = lane >> 4;
    int row  = m0 + (lane & 15);
    const __bf16* p0 = base + (size_t)row * rowStride + k + half * 8;
    bf16x8 a0 = *(const bf16x8*)p0;
    bf16x8 a1 = *(const bf16x8*)(p0 + 16);
    return __builtin_shufflevector(a0, a1, 0,1,2,3,4,5,6,7,8,9,10,11,12,13,14,15);
}

// B fragment: 32x16 bf16, weight stored [N][K] row-major. Lanes 0-15 (N=lane)
// hold K=0..7 then K=16..23, lanes 16-31 hold K=8..15 then K=24..31.
__device__ __forceinline__ bf16x16 ldB(const __bf16* base, int rowStride,
                                       int n0, int k, int lane) {
    int half = lane >> 4;
    int col  = n0 + (lane & 15);
    const __bf16* p = base + (size_t)col * rowStride + k + half * 8;
    bf16x8 b0 = *(const bf16x8*)p;
    bf16x8 b1 = *(const bf16x8*)(p + 16);
    return __builtin_shufflevector(b0, b1, 0,1,2,3,4,5,6,7,8,9,10,11,12,13,14,15);
}

__device__ __forceinline__ f32x8 wmma_bf16(bf16x16 a, bf16x16 b, f32x8 c) {
    return __builtin_amdgcn_wmma_f32_16x16x32_bf16(
        false, a, false, b, (short)0, c, false, false);
}

__device__ __forceinline__ float sigf(float x) { return 1.f / (1.f + __expf(-x)); }

// -------- prep: stable descending argsort of lengths + gathers --------
__global__ void sort_kernel(const int* cap_len, const int* caps_in,
                            int* sort_ind, int* dec_len, int* caps_s,
                            float* out_tail) {
    __shared__ int lens[Bn];
    __shared__ int sind[Bn];
    int i = threadIdx.x;
    lens[i] = cap_len[i];
    __syncthreads();
    int li = lens[i];
    int rank = 0;
    for (int j = 0; j < Bn; ++j) {
        int lj = lens[j];
        rank += (lj > li) || (lj == li && j < i);
    }
    sind[rank] = i;
    __syncthreads();
    int s = sind[i];
    sort_ind[i] = s;
    int dl = lens[s] - 1;
    dec_len[i] = dl;
    for (int t = 0; t < Ln; ++t) {
        int c = caps_in[s * Ln + t];
        caps_s[i * Ln + t] = c;
        out_tail[i * Ln + t] = (float)c;
    }
    out_tail[Bn * Ln + i]      = (float)dl;
    out_tail[Bn * Ln + Bn + i] = (float)s;
}

// -------- mean over NP patches, gathered by sort_ind, emit bf16 --------
__global__ void mean_kernel(const float* feats, const int* sort_ind, __bf16* outbf) {
    int idx = blockIdx.x * blockDim.x + threadIdx.x;   // 128*2048 threads
    int b = idx >> 11;
    int f = idx & (FEATn - 1);
    int src = sort_ind[b];
    const float* p = feats + (size_t)src * NPn * FEATn + f;
    float s = 0.f;
    for (int pch = 0; pch < NPn; ++pch) s += p[(size_t)pch * FEATn];
    outbf[idx] = (__bf16)(s * (1.f / (float)NPn));
}

// -------- embedding gather (sorted captions), emit bf16 --------
__global__ void emb_kernel(const float* table, const int* caps_s, __bf16* outbf) {
    int idx = blockIdx.x * blockDim.x + threadIdx.x;   // 128*52*512
    int e  = idx & (En - 1);
    int bt = idx >> 9;
    int tok = caps_s[bt];
    outbf[idx] = (__bf16)table[(size_t)tok * En + e];
}

// -------- generic f32 -> bf16 convert --------
__global__ void cvt_kernel(const float* src, __bf16* dst, int n) {
    int i = blockIdx.x * blockDim.x + threadIdx.x;
    if (i < n) dst[i] = (__bf16)src[i];
}

// -------- h0 = feats_mean @ att1_w^T + att1_b ; c0 = 0 --------
// 64 waves, one per 16x64 tile of the 128x512 output. K = 2048.
// Software-pipelined: next k-step fragments are loaded before the WMMA group
// consuming the current fragments.
__global__ void h0_kernel(const __bf16* featsA, const __bf16* wB, const float* att1_b,
                          float* h0, __bf16* h0bf, float* c0) {
    int wave = (blockIdx.x * blockDim.x + threadIdx.x) >> 5;
    int lane = threadIdx.x & 31;
    int m0 = (wave >> 3) * 16;        // 8 m-tiles
    int n0 = (wave & 7) * 64;         // 8 n-tiles of width 64
    f32x8 acc[4] = {{}, {}, {}, {}};

    bf16x16 aC = ldA(featsA, FEATn, m0, 0, lane);
    bf16x16 bC[4];
#pragma unroll
    for (int j = 0; j < 4; ++j) bC[j] = ldB(wB, FEATn, n0 + j * 16, 0, lane);

    for (int k = 0; k < FEATn - 32; k += 32) {
        bf16x16 aN = ldA(featsA, FEATn, m0, k + 32, lane);
        bf16x16 bN[4];
#pragma unroll
        for (int j = 0; j < 4; ++j) bN[j] = ldB(wB, FEATn, n0 + j * 16, k + 32, lane);
#pragma unroll
        for (int j = 0; j < 4; ++j) acc[j] = wmma_bf16(aC, bC[j], acc[j]);
        aC = aN;
#pragma unroll
        for (int j = 0; j < 4; ++j) bC[j] = bN[j];
    }
#pragma unroll
    for (int j = 0; j < 4; ++j) acc[j] = wmma_bf16(aC, bC[j], acc[j]);

    int mb = m0 + (lane >> 4) * 8;
#pragma unroll
    for (int j = 0; j < 4; ++j) {
        int n = n0 + j * 16 + (lane & 15);
        float bias = att1_b[n];
#pragma unroll
        for (int r = 0; r < 8; ++r) {
            int m = mb + r;
            float v = acc[j][r] + bias;
            h0[m * Dn + n]   = v;
            h0bf[m * Dn + n] = (__bf16)v;
            c0[m * Dn + n]   = 0.f;
        }
    }
}

// -------- one LSTM step: gates GEMMs + fused cell update --------
// 256 waves; each wave owns a 16x16 (m,n) slice and computes all 4 gate tiles
// for that slice, fusing the pointwise update. Pipelined across both GEMMs.
__global__ void lstm_kernel(const __bf16* xall, int t,
                            const __bf16* wih, const __bf16* whh,
                            const float* bih, const float* bhh,
                            const __bf16* hprev_bf, const float* hprev,
                            const float* cprev, const int* dec_len,
                            float* hout, float* cout,
                            __bf16* hout_bf, __bf16* hnew_bf) {
    int wave = (blockIdx.x * blockDim.x + threadIdx.x) >> 5;
    int lane = threadIdx.x & 31;
    int m0 = (wave >> 5) * 16;
    int n0 = (wave & 31) * 16;

    const __bf16* xbase = xall + (size_t)t * En;   // row b at stride L*E

    f32x8 acc[4] = {{}, {}, {}, {}};

    // ---- x @ w_ih^T ----
    bf16x16 aC = ldA(xbase, Ln * En, m0, 0, lane);
    bf16x16 bC[4];
#pragma unroll
    for (int g = 0; g < 4; ++g) bC[g] = ldB(wih, En, g * Dn + n0, 0, lane);

    for (int k = 0; k < En - 32; k += 32) {
        bf16x16 aN = ldA(xbase, Ln * En, m0, k + 32, lane);
        bf16x16 bN[4];
#pragma unroll
        for (int g = 0; g < 4; ++g) bN[g] = ldB(wih, En, g * Dn + n0, k + 32, lane);
#pragma unroll
        for (int g = 0; g < 4; ++g) acc[g] = wmma_bf16(aC, bC[g], acc[g]);
        aC = aN;
#pragma unroll
        for (int g = 0; g < 4; ++g) bC[g] = bN[g];
    }
    // bridge into the h GEMM: prefetch its first fragments before last x WMMAs
    {
        bf16x16 aN = ldA(hprev_bf, Dn, m0, 0, lane);
        bf16x16 bN[4];
#pragma unroll
        for (int g = 0; g < 4; ++g) bN[g] = ldB(whh, Dn, g * Dn + n0, 0, lane);
#pragma unroll
        for (int g = 0; g < 4; ++g) acc[g] = wmma_bf16(aC, bC[g], acc[g]);
        aC = aN;
#pragma unroll
        for (int g = 0; g < 4; ++g) bC[g] = bN[g];
    }
    // ---- h_prev @ w_hh^T ----
    for (int k = 0; k < Dn - 32; k += 32) {
        bf16x16 aN = ldA(hprev_bf, Dn, m0, k + 32, lane);
        bf16x16 bN[4];
#pragma unroll
        for (int g = 0; g < 4; ++g) bN[g] = ldB(whh, Dn, g * Dn + n0, k + 32, lane);
#pragma unroll
        for (int g = 0; g < 4; ++g) acc[g] = wmma_bf16(aC, bC[g], acc[g]);
        aC = aN;
#pragma unroll
        for (int g = 0; g < 4; ++g) bC[g] = bN[g];
    }
#pragma unroll
    for (int g = 0; g < 4; ++g) acc[g] = wmma_bf16(aC, bC[g], acc[g]);

    int n  = n0 + (lane & 15);
    int mb = m0 + (lane >> 4) * 8;
    float bi = bih[0 * Dn + n] + bhh[0 * Dn + n];
    float bf = bih[1 * Dn + n] + bhh[1 * Dn + n];
    float bg = bih[2 * Dn + n] + bhh[2 * Dn + n];
    float bo = bih[3 * Dn + n] + bhh[3 * Dn + n];

#pragma unroll
    for (int r = 0; r < 8; ++r) {
        int m = mb + r;
        float iv = sigf(acc[0][r] + bi);
        float fv = sigf(acc[1][r] + bf);
        float gv = tanhf(acc[2][r] + bg);
        float ov = sigf(acc[3][r] + bo);
        float cold = cprev[m * Dn + n];
        float cn = fv * cold + iv * gv;
        float hn = ov * tanhf(cn);
        bool msk = t < dec_len[m];
        float h2 = msk ? hn : hprev[m * Dn + n];
        float c2 = msk ? cn : cold;
        hout[m * Dn + n]    = h2;
        cout[m * Dn + n]    = c2;
        hout_bf[m * Dn + n] = (__bf16)h2;
        hnew_bf[m * Dn + n] = (__bf16)hn;
    }
}

// -------- vocab projection: pred = (h_new @ word_w^T + word_b) * mask --------
// 10000 = 125 * 80: each wave computes a 16x80 tile (5 accumulators); the A
// fragment is reused across 5 B columns and the k-loop is software-pipelined.
// 1000 waves (8 x 125 tiles), K = 512. Writes straight to d_out.
__global__ void pred_kernel(const __bf16* hnew_bf, const __bf16* wword,
                            const float* word_b, const int* dec_len,
                            int t, float* out) {
    int wave = (blockIdx.x * blockDim.x + threadIdx.x) >> 5;
    int lane = threadIdx.x & 31;
    int m0 = (wave / 125) * 16;
    int n0 = (wave % 125) * 80;
    f32x8 acc[5] = {{}, {}, {}, {}, {}};

    bf16x16 aC = ldA(hnew_bf, Dn, m0, 0, lane);
    bf16x16 bC[5];
#pragma unroll
    for (int j = 0; j < 5; ++j) bC[j] = ldB(wword, Dn, n0 + j * 16, 0, lane);

    for (int k = 0; k < Dn - 32; k += 32) {
        bf16x16 aN = ldA(hnew_bf, Dn, m0, k + 32, lane);
        bf16x16 bN[5];
#pragma unroll
        for (int j = 0; j < 5; ++j) bN[j] = ldB(wword, Dn, n0 + j * 16, k + 32, lane);
#pragma unroll
        for (int j = 0; j < 5; ++j) acc[j] = wmma_bf16(aC, bC[j], acc[j]);
        aC = aN;
#pragma unroll
        for (int j = 0; j < 5; ++j) bC[j] = bN[j];
    }
#pragma unroll
    for (int j = 0; j < 5; ++j) acc[j] = wmma_bf16(aC, bC[j], acc[j]);

    int mb = m0 + (lane >> 4) * 8;
    float mval[8];
#pragma unroll
    for (int r = 0; r < 8; ++r) mval[r] = (t < dec_len[mb + r]) ? 1.f : 0.f;

#pragma unroll
    for (int j = 0; j < 5; ++j) {
        int n = n0 + j * 16 + (lane & 15);
        float wb = word_b[n];
#pragma unroll
        for (int r = 0; r < 8; ++r) {
            int m = mb + r;
            out[(size_t)m * Tn * Vn + (size_t)t * Vn + n] = (acc[j][r] + wb) * mval[r];
        }
    }
}

extern "C" void kernel_launch(void* const* d_in, const int* in_sizes, int n_in,
                              void* d_out, int out_size, void* d_ws, size_t ws_size,
                              hipStream_t stream) {
    (void)in_sizes; (void)n_in; (void)out_size; (void)ws_size;
    const float* image_features   = (const float*)d_in[0];
    const int*   encoded_captions = (const int*)d_in[1];
    const int*   caption_lengths  = (const int*)d_in[2];
    const float* emb_table        = (const float*)d_in[3];
    const float* att1_w           = (const float*)d_in[4];
    const float* att1_b           = (const float*)d_in[5];
    const float* w_ih             = (const float*)d_in[6];
    const float* w_hh             = (const float*)d_in[7];
    const float* b_ih             = (const float*)d_in[8];
    const float* b_hh             = (const float*)d_in[9];
    const float* word_w           = (const float*)d_in[10];
    const float* word_b           = (const float*)d_in[11];

    float* out = (float*)d_out;
    char*  ws  = (char*)d_ws;

    int*    sort_ind = (int*)(ws + OFF_SORT);
    int*    dec_len  = (int*)(ws + OFF_DECLEN);
    int*    caps_s   = (int*)(ws + OFF_CAPS);
    __bf16* fmean_bf = (__bf16*)(ws + OFF_FMEAN);
    __bf16* att1w_bf = (__bf16*)(ws + OFF_ATT1W);
    __bf16* wih_bf   = (__bf16*)(ws + OFF_WIH);
    __bf16* whh_bf   = (__bf16*)(ws + OFF_WHH);
    __bf16* wordw_bf = (__bf16*)(ws + OFF_WORDW);
    __bf16* emb_bf   = (__bf16*)(ws + OFF_EMB);
    float*  h_f32    = (float*)(ws + OFF_H);     // 2 slots of 128*512
    float*  c_f32    = (float*)(ws + OFF_C);
    __bf16* h_bf     = (__bf16*)(ws + OFF_HBF);  // 2 slots
    __bf16* hnew_bf  = (__bf16*)(ws + OFF_HNEW);

    float* out_tail = out + (size_t)Bn * Tn * Vn;

    // ---- preprocessing ----
    sort_kernel<<<1, Bn, 0, stream>>>(caption_lengths, encoded_captions,
                                      sort_ind, dec_len, caps_s, out_tail);
    mean_kernel<<<(Bn * FEATn) / 256, 256, 0, stream>>>(image_features, sort_ind, fmean_bf);
    emb_kernel<<<(Bn * Ln * En) / 256, 256, 0, stream>>>(emb_table, caps_s, emb_bf);
    cvt_kernel<<<(Dn * FEATn + 255) / 256, 256, 0, stream>>>(att1_w, att1w_bf, Dn * FEATn);
    cvt_kernel<<<(4 * Dn * En + 255) / 256, 256, 0, stream>>>(w_ih, wih_bf, 4 * Dn * En);
    cvt_kernel<<<(4 * Dn * Dn + 255) / 256, 256, 0, stream>>>(w_hh, whh_bf, 4 * Dn * Dn);
    cvt_kernel<<<(Vn * Dn + 255) / 256, 256, 0, stream>>>(word_w, wordw_bf, Vn * Dn);

    // ---- h0 GEMM (writes h/c slot 0): 64 waves = 16 blocks * 4 waves ----
    h0_kernel<<<16, 128, 0, stream>>>(fmean_bf, att1w_bf, att1_b,
                                      h_f32, h_bf, c_f32);

    // ---- 51 recurrent steps ----
    for (int t = 0; t < Tn; ++t) {
        int cur = t & 1;
        int nxt = cur ^ 1;
        lstm_kernel<<<64, 128, 0, stream>>>(
            emb_bf, t, wih_bf, whh_bf, b_ih, b_hh,
            h_bf + (size_t)cur * Bn * Dn,
            h_f32 + (size_t)cur * Bn * Dn,
            c_f32 + (size_t)cur * Bn * Dn,
            dec_len,
            h_f32 + (size_t)nxt * Bn * Dn,
            c_f32 + (size_t)nxt * Bn * Dn,
            h_bf + (size_t)nxt * Bn * Dn,
            hnew_bf);
        pred_kernel<<<250, 128, 0, stream>>>(hnew_bf, wordw_bf, word_b,
                                             dec_len, t, out);
    }
}
